// AudioXMMDiTSelfAttention_61881888801240
// MI455X (gfx1250) — matrix operand order repaired and checked by reading, compiled, not run on hardware
//
#include <hip/hip_runtime.h>

// ---------------------------------------------------------------------------
// MI455X (gfx1250) fused self-attention:
//   qkv = x @ Wqkv^T + b  ->  rmsnorm(q,k) -> rope(q,k) -> flash attention
// Precision: bf16 WMMA (v_wmma_f32_16x16x32_bf16) with fp32 accumulation.
// Fragments are wide b128 loads (fragment-major LDS tiles, V transposed).
// Flash attention stages K/V via the Tensor Data Mover (tensor_load_to_lds)
// with double buffering: DMA of block kb+1 overlaps WMMA on block kb.
// Workspace: q/k [B,H,N,128] bf16, v [B,H,128,N] bf16 = 48 MB in d_ws.
// ---------------------------------------------------------------------------

typedef unsigned short ushort_t;
typedef unsigned int   uint_t;
typedef unsigned long long u64_t;
typedef __bf16 bf16_t;
typedef __bf16 v16bf __attribute__((ext_vector_type(16)));
typedef float  v8f   __attribute__((ext_vector_type(8)));
typedef unsigned int u32x4 __attribute__((ext_vector_type(4)));
typedef unsigned int u32x8 __attribute__((ext_vector_type(8)));

struct U4x2 { uint4 a, b; };   // 32 bytes, bit-castable to v16bf

#define DIM      2048
#define NHEADS   16
#define HEAD_DIM 128
#define BATCH    2
#define SEQ      2048
#define MTOT     (BATCH * SEQ)   // 4096 rows of x
#define NOUT     (3 * DIM)       // 6144 qkv columns

__device__ __forceinline__ ushort_t f2bf(float f) {
  return __builtin_bit_cast(ushort_t, (bf16_t)f);   // native v_cvt f32->bf16
}
__device__ __forceinline__ uint_t pack2(float a, float b) {
#if __has_builtin(__builtin_amdgcn_cvt_pk_bf16_f32)
  typedef __bf16 bf2v __attribute__((ext_vector_type(2)));
  bf2v t = __builtin_amdgcn_cvt_pk_bf16_f32(a, b);
  return __builtin_bit_cast(uint_t, t);
#else
  return (uint_t)f2bf(a) | ((uint_t)f2bf(b) << 16);
#endif
}
__device__ __forceinline__ float bf2f(ushort_t h) {
  return __builtin_bit_cast(float, (uint_t)h << 16);
}
__device__ __forceinline__ v16bf ldsFrag(const ushort_t* p) {
  return *(const v16bf*)p;                          // 32B -> 2x ds_load_b128
}
__device__ __forceinline__ v16bf joinFrag(uint4 a, uint4 b) {
  U4x2 u{a, b};
  return __builtin_bit_cast(v16bf, u);
}
__device__ __forceinline__ uint_t lds_off(const void* p) {
  return (uint_t)(u64_t)p;    // LDS aperture keeps offset in low 32 bits
}

// ---------------------------------------------------------------------------
// Tensor Data Mover: 2D tile load global->LDS (CDNA5 ISA)
// data_size = 2 bytes; dims/strides in elements.  Tracked with TENSORcnt.
// ---------------------------------------------------------------------------
__device__ __forceinline__ void tdm_load_2d(
    uint_t lds_addr, const void* gptr,
    uint_t tensor_d0, uint_t tensor_d1, u64_t stride0,
    uint_t tile_d0, uint_t tile_d1)
{
  const u64_t ga = (u64_t)gptr;
  u32x4 g0;
  g0[0] = 1u;                                   // count=1, user descriptor
  g0[1] = lds_addr;                             // LDS byte address
  g0[2] = (uint_t)ga;                           // global_addr[31:0]
  g0[3] = (uint_t)(ga >> 32) | (2u << 30);      // global_addr[56:32] | type=2
  u32x8 g1;
  g1[0] = 1u << 16;                             // workgroup_mask=0, data_size=2B
  g1[1] = (tensor_d0 & 0xFFFFu) << 16;          // tensor_dim0[15:0]
  g1[2] = (tensor_d0 >> 16) | ((tensor_d1 & 0xFFFFu) << 16);
  g1[3] = (tensor_d1 >> 16) | (tile_d0 << 16);  // tile_dim0
  g1[4] = tile_d1;                              // tile_dim1 (tile_dim2 = 0)
  g1[5] = (uint_t)stride0;                      // tensor_dim0_stride[31:0]
  g1[6] = (uint_t)(stride0 >> 32);              // stride0[47:32], stride1=0
  g1[7] = 0u;
  asm volatile("tensor_load_to_lds %0, %1" :: "s"(g0), "s"(g1) : "memory");
}

// ---------------------------------------------------------------------------
// Kernel 1: qkv GEMM.  C[4096,6144] = X[4096,2048] @ W^T + bias.
// LDS tiles stored fragment-major: tile[frag16][lane][elem16] so a wave
// fetches a whole 16x32 bf16 fragment with one 32-byte LDS access.
// Epilogue de-interleaves j = h*384 + d*3 + c into q/k ([B,H,N,D]) and
// v transposed ([B,H,D,N]); j-decode hoisted out of the row loops.
// ---------------------------------------------------------------------------
#define TM 128
#define TN 128
#define TK 32

__global__ __launch_bounds__(256) void qkv_gemm_kernel(
    const float* __restrict__ x, const float* __restrict__ w,
    const float* __restrict__ bias,
    ushort_t* __restrict__ qbuf, ushort_t* __restrict__ kbuf,
    ushort_t* __restrict__ vbuf)
{
  // [8 frag-blocks][32 lanes][16 elems] bf16 = 8 KB each
  __shared__ __attribute__((aligned(32))) ushort_t As2[8 * 32 * 16];
  __shared__ __attribute__((aligned(32))) ushort_t Bs2[8 * 32 * 16];

  const int tid  = threadIdx.x;
  const int lane = tid & 31;
  const int wave = tid >> 5;
  const int lm   = lane & 15;
  const int lh   = lane >> 4;
  const int wm   = wave >> 1;   // 0..3 -> 32-row slice
  const int wn   = wave & 1;    // 0..1 -> 64-col slice

  const int jBase   = blockIdx.x * TN;
  const int rowBase = blockIdx.y * TM;

  v8f acc[2][4] = {};

  for (int k0 = 0; k0 < DIM; k0 += TK) {
    // ---- stage A (128x32 f32 -> bf16, fragment-major swizzle)
#pragma unroll
    for (int it = 0; it < 4; ++it) {
      const int fi = tid + it * 256;            // 0..1023 float4 tiles
      const int m  = fi >> 3;
      const int kq = fi & 7;                    // k = kq*4 .. kq*4+3
      const float4 v = *(const float4*)(x + (size_t)(rowBase + m) * DIM + k0 + kq * 4);
      // A-frag layout: elem e<8 -> k=8*lh+e ; e>=8 -> k=16+8*lh+(e-8)
      const int lhh = (kq >> 1) & 1;
      const int e0  = (kq & 1) * 4 + (kq >> 2) * 8;
      uint2 pk; pk.x = pack2(v.x, v.y); pk.y = pack2(v.z, v.w);
      *(uint2*)&As2[(m >> 4) * 512 + (lhh * 16 + (m & 15)) * 16 + e0] = pk;
    }
    // ---- stage B (W rows contiguous in k; B-frag layout: elem e<8 -> k=8*lh+e ; e>=8 -> k=16+8*lh+(e-8))
#pragma unroll
    for (int it = 0; it < 4; ++it) {
      const int fi = tid + it * 256;
      const int j  = fi >> 3;
      const int kq = fi & 7;
      const float4 v = *(const float4*)(w + (size_t)(jBase + j) * DIM + k0 + kq * 4);
      const int lhh = (kq >> 1) & 1;
      const int e0  = (kq & 1) * 4 + (kq >> 2) * 8;
      uint2 pk; pk.x = pack2(v.x, v.y); pk.y = pack2(v.z, v.w);
      *(uint2*)&Bs2[(j >> 4) * 512 + (lhh * 16 + (j & 15)) * 16 + e0] = pk;
    }
    __syncthreads();

    v16bf af[2];
#pragma unroll
    for (int mi = 0; mi < 2; ++mi)
      af[mi] = ldsFrag(&As2[(wm * 2 + mi) * 512 + lane * 16]);
#pragma unroll
    for (int ni = 0; ni < 4; ++ni) {
      const v16bf bfrag = ldsFrag(&Bs2[(wn * 4 + ni) * 512 + lane * 16]);
#pragma unroll
      for (int mi = 0; mi < 2; ++mi)
        acc[mi][ni] = __builtin_amdgcn_wmma_f32_16x16x32_bf16(
            false, af[mi], false, bfrag, (short)0, acc[mi][ni], false, false);
    }
    __syncthreads();
  }

  // ---- epilogue: + bias, de-interleave j = h*384 + d*3 + c (hoisted per ni)
  const int bb     = rowBase >> 11;            // batch (block-uniform)
  const int nqBase = (rowBase & (SEQ - 1)) + wm * 32 + 8 * lh;
#pragma unroll
  for (int ni = 0; ni < 4; ++ni) {
    const int j  = jBase + wn * 64 + ni * 16 + lm;
    const int h  = j / 384;
    const int rr = j - h * 384;
    const int d  = rr / 3;
    const int c  = rr - d * 3;
    const float bj = bias[j];
    const int bh = bb * NHEADS + h;
    if (c == 2) {                               // V transposed: [B,H,D,N]
      ushort_t* vrow = vbuf + ((size_t)bh * HEAD_DIM + d) * SEQ;
#pragma unroll
      for (int mi = 0; mi < 2; ++mi)
#pragma unroll
        for (int r = 0; r < 8; ++r)
          vrow[nqBase + mi * 16 + r] = f2bf(acc[mi][ni][r] + bj);
    } else {
      ushort_t* dst = ((c == 0) ? qbuf : kbuf) + (size_t)bh * SEQ * HEAD_DIM + d;
#pragma unroll
      for (int mi = 0; mi < 2; ++mi)
#pragma unroll
        for (int r = 0; r < 8; ++r)
          dst[(size_t)(nqBase + mi * 16 + r) * HEAD_DIM] = f2bf(acc[mi][ni][r] + bj);
    }
  }
}

// ---------------------------------------------------------------------------
// Kernel 2: RMSNorm + interleaved RoPE on q and k, in place (bf16 <-> fp32).
// One wave per 128-element head row; 4 elems (2 rope pairs) per lane.
// ---------------------------------------------------------------------------
__global__ __launch_bounds__(256) void norm_rope_kernel(
    ushort_t* __restrict__ qbuf, ushort_t* __restrict__ kbuf,
    const float* __restrict__ cosb, const float* __restrict__ sinb)
{
  const int wave = threadIdx.x >> 5;
  const int lane = threadIdx.x & 31;
  const int row  = blockIdx.x * 8 + wave;              // 0 .. 2*B*H*N-1
  const int nrows = BATCH * NHEADS * SEQ;              // 65536
  ushort_t* base = (row < nrows) ? qbuf : kbuf;
  const int r  = row & (nrows - 1);
  const int nq = r & (SEQ - 1);
  ushort_t* p = base + (size_t)r * HEAD_DIM + lane * 4;

  const uint_t u0 = *(const uint_t*)p;
  const uint_t u1 = *(const uint_t*)(p + 2);
  float e0 = bf2f((ushort_t)u0), e1 = bf2f((ushort_t)(u0 >> 16));
  float e2 = bf2f((ushort_t)u1), e3 = bf2f((ushort_t)(u1 >> 16));

  float ss = e0 * e0 + e1 * e1 + e2 * e2 + e3 * e3;
#pragma unroll
  for (int o = 16; o >= 1; o >>= 1) ss += __shfl_xor(ss, o, 32);
  const float rms = rsqrtf(ss * (1.0f / HEAD_DIM) + 1e-6f);

  const int pi = lane * 2;
  const float c0 = cosb[nq * 64 + pi],     s0 = sinb[nq * 64 + pi];
  const float c1 = cosb[nq * 64 + pi + 1], s1 = sinb[nq * 64 + pi + 1];
  e0 *= rms; e1 *= rms; e2 *= rms; e3 *= rms;
  *(uint_t*)p       = pack2(e0 * c0 - e1 * s0, e1 * c0 + e0 * s0);
  *(uint_t*)(p + 2) = pack2(e2 * c1 - e3 * s1, e3 * c1 + e2 * s1);
}

// ---------------------------------------------------------------------------
// Kernel 3: flash attention per (b,h).  WG = 8 waves, 16 q rows per wave
// (128 q rows per WG).  64-key blocks double-buffered in LDS, filled by the
// Tensor Data Mover: wave 0 issues tensor_load_to_lds for block kb+1, waits
// tensorcnt<=2 (block kb complete), so DMA overlaps the WMMA work.
// Online softmax in fp32; P transposed to A-layout via per-wave LDS buffer.
// ---------------------------------------------------------------------------
__global__ __launch_bounds__(256) void flash_attn_kernel(
    const ushort_t* __restrict__ qbuf, const ushort_t* __restrict__ kbuf,
    const ushort_t* __restrict__ vbuf, float* __restrict__ out)
{
  __shared__ __attribute__((aligned(32))) ushort_t Ks[2][64 * HEAD_DIM]; // [key][d]  32 KB
  __shared__ __attribute__((aligned(32))) ushort_t Vt[2][HEAD_DIM * 64]; // [d][key]  32 KB
  __shared__ __attribute__((aligned(32))) ushort_t Ps[8][16 * 64];       // per-wave  16 KB

  const int bh   = blockIdx.y;            // 0..31
  const int b    = bh >> 4;
  const int h    = bh & 15;
  const int wave = threadIdx.x >> 5;
  const int lane = threadIdx.x & 31;
  const int lm   = lane & 15;
  const int lh   = lane >> 4;
  const int qblk = blockIdx.x;            // 0..15
  const int NB   = SEQ / 64;              // 32 key blocks

  const ushort_t* Qh = qbuf + (size_t)bh * SEQ * HEAD_DIM;
  const ushort_t* Kh = kbuf + (size_t)bh * SEQ * HEAD_DIM;
  const ushort_t* Vh = vbuf + (size_t)bh * HEAD_DIM * SEQ;   // transposed [d][n]

  const int qRow0 = qblk * 128 + wave * 16;
  const float scale = 0.08838834764831845f;  // 1/sqrt(128)

  // Q A-fragments over d: 4 frags of 16x32 (two 16B runs each)
  v16bf qf[4];
  {
    const ushort_t* qrow = Qh + (size_t)(qRow0 + lm) * HEAD_DIM;
#pragma unroll
    for (int kf = 0; kf < 4; ++kf) {
      const uint4 lo = *(const uint4*)(qrow + kf * 32 + 16 * lh);
      const uint4 hi = *(const uint4*)(qrow + kf * 32 + 16 * lh + 8);
      qf[kf] = joinFrag(lo, hi);
    }
  }

  // TDM issue of one 64-key block (K row-major tile + V transposed tile)
  auto issue_block = [&](int blk) {
    const int buf = blk & 1;
    tdm_load_2d(lds_off(&Ks[buf][0]), Kh + (size_t)blk * 64 * HEAD_DIM,
                /*tensor_d0=*/HEAD_DIM, /*tensor_d1=*/SEQ, /*stride0=*/HEAD_DIM,
                /*tile_d0=*/HEAD_DIM, /*tile_d1=*/64);
    tdm_load_2d(lds_off(&Vt[buf][0]), Vh + (size_t)blk * 64,
                /*tensor_d0=*/SEQ, /*tensor_d1=*/HEAD_DIM, /*stride0=*/SEQ,
                /*tile_d0=*/64, /*tile_d1=*/HEAD_DIM);
  };
  if (wave == 0) issue_block(0);

  v8f oacc[8] = {};
  float rowM[8], rowL[8];
#pragma unroll
  for (int r = 0; r < 8; ++r) { rowM[r] = -1e30f; rowL[r] = 0.0f; }

  for (int kb = 0; kb < NB; ++kb) {
    const int buf = kb & 1;
    if (wave == 0) {
      if (kb + 1 < NB) {
        issue_block(kb + 1);                       // prefetch next block
        __builtin_amdgcn_s_wait_tensorcnt(2);      // block kb complete
      } else {
        __builtin_amdgcn_s_wait_tensorcnt(0);
      }
    }
    __syncthreads();                               // buffer kb visible to all

    // ---- S = Q . K^T  (16 q rows x 64 keys): B-frag = contiguous run of Ks row
    v8f s[4];
#pragma unroll
    for (int nf = 0; nf < 4; ++nf) {
      v8f a = {};
      const ushort_t* krow = &Ks[buf][(size_t)(nf * 16 + lm) * HEAD_DIM];
#pragma unroll
      for (int kf = 0; kf < 4; ++kf) {
        const v16bf bfrag = ldsFrag(krow + kf * 32 + lh * 16);
        a = __builtin_amdgcn_wmma_f32_16x16x32_bf16(
            false, qf[kf], false, bfrag, (short)0, a, false, false);
      }
      s[nf] = a;
    }

    // ---- online softmax (row m = r + 8*lh; row data across 16 lanes)
    float alpha[8];
#pragma unroll
    for (int r = 0; r < 8; ++r) {
      float t = fmaxf(fmaxf(s[0][r], s[1][r]), fmaxf(s[2][r], s[3][r])) * scale;
#pragma unroll
      for (int o = 8; o >= 1; o >>= 1) t = fmaxf(t, __shfl_xor(t, o, 32));
      const float mnew = fmaxf(rowM[r], t);
      alpha[r] = __expf(rowM[r] - mnew);
      rowM[r]  = mnew;
    }
#pragma unroll
    for (int r = 0; r < 8; ++r) {
      float ps = 0.0f;
#pragma unroll
      for (int nf = 0; nf < 4; ++nf) {
        const float pv = __expf(s[nf][r] * scale - rowM[r]);
        Ps[wave][(r + 8 * lh) * 64 + nf * 16 + lm] = f2bf(pv);
        ps += pv;
      }
#pragma unroll
      for (int o = 8; o >= 1; o >>= 1) ps += __shfl_xor(ps, o, 32);
      rowL[r] = rowL[r] * alpha[r] + ps;
    }
#pragma unroll
    for (int nfo = 0; nfo < 8; ++nfo)
#pragma unroll
      for (int r = 0; r < 8; ++r) oacc[nfo][r] *= alpha[r];

    // wave-local LDS store->load ordering for the P transpose
    asm volatile("s_wait_dscnt 0" ::: "memory");

    // ---- P A-fragments (16 x 64 keys -> two 16x32 frags, two 16B runs each)
    v16bf pf[2];
#pragma unroll
    for (int kf = 0; kf < 2; ++kf) {
      const ushort_t* prow = &Ps[wave][lm * 64 + kf * 32];
      const uint4 lo = *(const uint4*)(prow + 16 * lh);
      const uint4 hi = *(const uint4*)(prow + 16 * lh + 8);
      pf[kf] = joinFrag(lo, hi);
    }

    // ---- O += P . V   (8 d-frags x 2 k-frags): B-frag = contiguous run of Vt row
#pragma unroll
    for (int nfo = 0; nfo < 8; ++nfo) {
      const ushort_t* vrow = &Vt[buf][(size_t)(nfo * 16 + lm) * 64];
#pragma unroll
      for (int kf = 0; kf < 2; ++kf) {
        const v16bf vfrag = ldsFrag(vrow + kf * 32 + lh * 16);
        oacc[nfo] = __builtin_amdgcn_wmma_f32_16x16x32_bf16(
            false, pf[kf], false, vfrag, (short)0, oacc[nfo], false, false);
      }
    }
    __syncthreads();    // all reads of buffer kb done before TDM refills it
  }

  // ---- epilogue: O /= l, scatter to out[b][nq][h*128+d] (fp32)
#pragma unroll
  for (int nfo = 0; nfo < 8; ++nfo)
#pragma unroll
    for (int r = 0; r < 8; ++r) {
      const int nq = qRow0 + r + 8 * lh;
      const int d  = nfo * 16 + lm;
      out[((size_t)(b * SEQ + nq)) * DIM + h * HEAD_DIM + d] = oacc[nfo][r] / rowL[r];
    }
}

// ---------------------------------------------------------------------------
extern "C" void kernel_launch(void* const* d_in, const int* in_sizes, int n_in,
                              void* d_out, int out_size, void* d_ws, size_t ws_size,
                              hipStream_t stream)
{
  (void)in_sizes; (void)n_in; (void)out_size; (void)ws_size;

  const float* x    = (const float*)d_in[0];
  const float* wqkv = (const float*)d_in[1];
  const float* bqkv = (const float*)d_in[2];
  const float* cosb = (const float*)d_in[3];
  const float* sinb = (const float*)d_in[4];
  float* out = (float*)d_out;

  const size_t headElems = (size_t)BATCH * NHEADS * SEQ * HEAD_DIM;  // 8 Mi elems
  ushort_t* qws = (ushort_t*)d_ws;            // 16 MB  [B,H,N,D]
  ushort_t* kws = qws + headElems;            // 16 MB  [B,H,N,D]
  ushort_t* vws = kws + headElems;            // 16 MB  [B,H,D,N] (transposed)

  qkv_gemm_kernel<<<dim3(NOUT / TN, MTOT / TM), 256, 0, stream>>>(
      x, wqkv, bqkv, qws, kws, vws);

  const int rows = 2 * BATCH * NHEADS * SEQ;  // q and k rows
  norm_rope_kernel<<<dim3(rows / 8), 256, 0, stream>>>(qws, kws, cosb, sinb);

  flash_attn_kernel<<<dim3(SEQ / 128, BATCH * NHEADS), 256, 0, stream>>>(
      qws, kws, vws, out);
}
